// RNNDecoder_5643587027293
// MI455X (gfx1250) — matrix operand order repaired and checked by reading, not compiled, not run on hardware
//
#include <hip/hip_runtime.h>

// Problem constants (from the reference)
#define BB   32
#define NN   512
#define TT   12
#define DD   2
#define HH   128
#define KHOP 2
#define MTOT (BB*NN)   // 16384 rows for the dense GEMMs

typedef __attribute__((ext_vector_type(16))) __bf16 v16bf;
typedef __attribute__((ext_vector_type(8)))  float  v8f;

union Frag { v16bf v; uint4 q[2]; };

static __device__ inline unsigned short f2bf(float f) {
    __bf16 b = (__bf16)f;                 // native cvt (RNE)
    unsigned short u; __builtin_memcpy(&u, &b, 2);
    return u;
}
static __device__ inline float bf2f(unsigned short h) {
    __bf16 b; __builtin_memcpy(&b, &h, 2);
    return (float)b;
}

// ---------------------------------------------------------------------------
// Tensor Data Mover: async 2D tile DMA global(bf16,row stride lda) -> LDS with
// padding so each 32-element (64B) row lands on an 80B-stride LDS row.
// ---------------------------------------------------------------------------
#if __has_builtin(__builtin_amdgcn_tensor_load_to_lds)
#define USE_TDM 1
typedef __attribute__((ext_vector_type(4))) unsigned int u32x4;
typedef __attribute__((ext_vector_type(8))) int i32x8;
typedef __attribute__((ext_vector_type(4))) int i32x4;

static __device__ inline void tdm_load_tile_a(const unsigned short* gsrc,
                                              unsigned lds_off, int lda) {
    unsigned long long ga = (unsigned long long)gsrc;
    u32x4 g0;
    g0[0] = 1u;                                   // count=1 (valid), user mode
    g0[1] = lds_off;                              // lds_addr (bytes)
    g0[2] = (unsigned)(ga & 0xFFFFFFFFu);         // global_addr[31:0]
    g0[3] = (unsigned)((ga >> 32) & 0x1FFFFFFu)   // global_addr[56:32]
          | (2u << 30);                           // type = 2 ("image")
    i32x8 g1;
    g1[0] = (1 << 16)        // data_size = 1 -> 2 bytes
          | (1 << 20)        // pad_enable
          | (3 << 22)        // pad_interval: 16 DWORDs (64B) before each pad
          | (3 << 25);       // pad_amount:   4 DWORDs (16B) -> 80B row stride
    g1[1] = (int)(32u << 16);          // tensor_dim0[15:0]=32 (== tile width)
    g1[2] = (int)(128u << 16);         // tensor_dim0 hi=0 | tensor_dim1 lo=128
    g1[3] = (int)(32u << 16);          // tensor_dim1 hi=0 | tile_dim0=32
    g1[4] = 128;                       // tile_dim1=128 | tile_dim2=0
    g1[5] = lda;                       // tensor_dim0_stride (elements), low 32
    g1[6] = 0;                         // stride hi | tensor_dim1_stride lo
    g1[7] = 0;
    i32x4 z4 = {};
#if __clang_major__ >= 23
    i32x8 z8 = {};
    __builtin_amdgcn_tensor_load_to_lds(g0, g1, z4, z4, z8, 0);
#else
    __builtin_amdgcn_tensor_load_to_lds(g0, g1, z4, z4, 0);
#endif
}
#else
#define USE_TDM 0
#endif

// ---------------------------------------------------------------------------
// Shared GEMM machinery: block tile 128(M) x 64(N), K-chunk 32, bf16 operands,
// fp32 WMMA accumulation. 256 threads = 8 waves; each wave owns a 32x32 tile.
// LDS row stride 40 ushorts (80B): 16B aligned, conflict-friendly, and equal
// to the TDM pad pattern above.
// ---------------------------------------------------------------------------
#define LDS_STRIDE 40

static __device__ inline void stage_a_manual(
    const unsigned short* __restrict__ Asrc, int lda, int m0, int k0,
    unsigned short* sA)
{
    for (int i = threadIdx.x; i < 128 * 16; i += 256) {
        int row = i >> 4, dw = i & 15;
        unsigned int v = *(const unsigned int*)(Asrc + (size_t)(m0 + row) * lda + k0 + dw * 2);
        *(unsigned int*)&sA[row * LDS_STRIDE + dw * 2] = v;
    }
}

static __device__ inline void stage_b(
    const unsigned short* __restrict__ Bsrc, int ldb, int k0, int n0,
    unsigned short* sB, int kNext, int K)
{
    // B tile: 32 k x 64 n, stored transposed (per-column contiguous K)
    for (int i = threadIdx.x; i < 32 * 32; i += 256) {
        int kk = i >> 5, np = i & 31;
        unsigned int v = *(const unsigned int*)(Bsrc + (size_t)(k0 + kk) * ldb + n0 + np * 2);
        sB[(np * 2 + 0) * LDS_STRIDE + kk] = (unsigned short)(v & 0xFFFFu);
        sB[(np * 2 + 1) * LDS_STRIDE + kk] = (unsigned short)(v >> 16);
        if (kNext < K)  // prefetch next K-panel of B (global_prefetch_b8)
            __builtin_prefetch(Bsrc + (size_t)(kNext + kk) * ldb + n0 + np * 2, 0, 1);
    }
}

static __device__ inline void wmma_step(const unsigned short* sA, const unsigned short* sB,
                                        int wm, int wn, v8f acc[2][2])
{
    const int lane = threadIdx.x & 31;
    const int cl = lane & 15, kh = lane >> 4;
    Frag a[2], b[2];
#pragma unroll
    for (int ai = 0; ai < 2; ++ai) {
        const uint4* p = (const uint4*)&sA[(wm + ai * 16 + cl) * LDS_STRIDE + kh * 8];
        a[ai].q[0] = p[0];   // K = kh*8 .. kh*8+7
        a[ai].q[1] = p[2];   // K = 16+kh*8 .. 16+kh*8+7
    }
#pragma unroll
    for (int bi = 0; bi < 2; ++bi) {
        const uint4* p = (const uint4*)&sB[(wn + bi * 16 + cl) * LDS_STRIDE + kh * 8];
        b[bi].q[0] = p[0];   // K = kh*8 .. kh*8+7
        b[bi].q[1] = p[2];   // K = 16+kh*8 .. 16+kh*8+7
    }
#pragma unroll
    for (int ai = 0; ai < 2; ++ai)
#pragma unroll
        for (int bi = 0; bi < 2; ++bi)
            acc[ai][bi] = __builtin_amdgcn_wmma_f32_16x16x32_bf16(
                false, a[ai].v, false, b[bi].v, (short)0, acc[ai][bi], false, false);
}

// Stage one K-chunk (A via TDM when available, B manual+transpose), then sync.
static __device__ inline void stage_chunk(
    const unsigned short* __restrict__ Asrc, int lda, int m0, int k0,
    const unsigned short* __restrict__ Bsrc, int ldb, int n0, int K,
    unsigned short* sA, unsigned short* sB)
{
#if USE_TDM
    if (threadIdx.x == 0)   // one wave issues the DMA (TDM ignores EXEC)
        tdm_load_tile_a(Asrc + (size_t)m0 * lda + k0,
                        (unsigned)(unsigned long long)(void*)sA, lda);
#else
    stage_a_manual(Asrc, lda, m0, k0, sA);
#endif
    stage_b(Bsrc, ldb, k0, n0, sB, k0 + 32, K);
#if USE_TDM
    __builtin_amdgcn_s_wait_tensorcnt(0);
#endif
    __syncthreads();
}

// ---------------------------------------------------------------------------
// GEMM 1: P[hop] = Z @ W[hop]     (M=16384, K in {160,256}, O in {128,256})
// hop = blockIdx.z in [0,3). Output stored as bf16 (it feeds GEMM 2).
// ---------------------------------------------------------------------------
__global__ __launch_bounds__(256) void k_gemm_zw(
    const unsigned short* __restrict__ Z, const unsigned short* __restrict__ W,
    unsigned short* __restrict__ P, int K, int O)
{
    __shared__ __align__(16) unsigned short sA[128 * LDS_STRIDE];
    __shared__ __align__(16) unsigned short sB[64 * LDS_STRIDE];

    const int hop = blockIdx.z;
    const unsigned short* Wb = W + (size_t)hop * K * O;
    unsigned short* Pb = P + (size_t)hop * MTOT * O;
    const int m0 = blockIdx.x * 128, n0 = blockIdx.y * 64;
    const int wid = threadIdx.x >> 5;
    const int wm = (wid & 3) * 32, wn = (wid >> 2) * 32;

    v8f acc[2][2] = {};
    for (int k0 = 0; k0 < K; k0 += 32) {
        stage_chunk(Z, K, m0, k0, Wb, O, n0, K, sA, sB);
        wmma_step(sA, sB, wm, wn, acc);
        __syncthreads();
    }
    const int lane = threadIdx.x & 31;
    const int cl = lane & 15, kh = lane >> 4;
#pragma unroll
    for (int ai = 0; ai < 2; ++ai)
#pragma unroll
        for (int bi = 0; bi < 2; ++bi)
#pragma unroll
            for (int j = 0; j < 8; ++j) {
                int r = m0 + wm + ai * 16 + kh * 8 + j;
                int c = n0 + wn + bi * 16 + cl;
                Pb[(size_t)r * O + c] = f2bf(acc[ai][bi][j]);
            }
}

// ---------------------------------------------------------------------------
// GEMM 2 (diffusion + fused GRU epilogue), per batch b = blockIdx.z:
//   acc = sum_s A_bf[s] @ P[s+1][b]        (M=512, K=2x512, N=O)
//   v   = acc + P[0][b] + bias
//   mode 0: RU = sigmoid(v)                              (O = 2H = 256)
//   mode 1: H  = u*H + (1-u)*tanh(v), u = RU[:, H+c]     (O = H = 128)
// ---------------------------------------------------------------------------
__global__ __launch_bounds__(256) void k_gemm_diff(
    const unsigned short* __restrict__ Abf,   // [2,512,512] bf16
    const unsigned short* __restrict__ P,     // [3, MTOT, O] bf16
    const float* __restrict__ bias,           // [O]
    float* __restrict__ RU,                   // [MTOT, 256]
    float* __restrict__ Hbuf,                 // [MTOT, 128]
    int O, int mode)
{
    __shared__ __align__(16) unsigned short sA[128 * LDS_STRIDE];
    __shared__ __align__(16) unsigned short sB[64 * LDS_STRIDE];

    const int b = blockIdx.z;
    const int m0 = blockIdx.x * 128, n0 = blockIdx.y * 64;
    const int wid = threadIdx.x >> 5;
    const int wm = (wid & 3) * 32, wn = (wid >> 2) * 32;

    v8f acc[2][2] = {};
    for (int s = 0; s < KHOP; ++s) {
        const unsigned short* Asrc = Abf + (size_t)s * NN * NN;
        const unsigned short* Bsrc = P + (size_t)(s + 1) * MTOT * O + (size_t)b * NN * O;
        for (int k0 = 0; k0 < NN; k0 += 32) {
            stage_chunk(Asrc, NN, m0, k0, Bsrc, O, n0, NN, sA, sB);
            wmma_step(sA, sB, wm, wn, acc);
            __syncthreads();
        }
    }
    const int lane = threadIdx.x & 31;
    const int cl = lane & 15, kh = lane >> 4;
#pragma unroll
    for (int ai = 0; ai < 2; ++ai)
#pragma unroll
        for (int bi = 0; bi < 2; ++bi)
#pragma unroll
            for (int j = 0; j < 8; ++j) {
                int rl = m0 + wm + ai * 16 + kh * 8 + j;   // row within batch (0..511)
                int c  = n0 + wn + bi * 16 + cl;
                size_t gm = (size_t)b * NN + rl;
                float v = acc[ai][bi][j] + bf2f(P[gm * O + c]) + bias[c];
                if (mode == 0) {
                    RU[gm * 256 + c] = 1.0f / (1.0f + __expf(-v));
                } else {
                    float u = RU[gm * 256 + HH + c];
                    float h = Hbuf[gm * HH + c];
                    Hbuf[gm * HH + c] = u * h + (1.0f - u) * tanhf(v);
                }
            }
}

// ---------------------------------------------------------------------------
// Elementwise / prep kernels
// ---------------------------------------------------------------------------
__global__ void k_f32copy(float* __restrict__ d, const float* __restrict__ s, int n) {
    int i = blockIdx.x * blockDim.x + threadIdx.x;
    if (i < n) d[i] = s[i];
}
__global__ void k_f2bf(unsigned short* __restrict__ d, const float* __restrict__ s, int n) {
    int i = blockIdx.x * blockDim.x + threadIdx.x;
    if (i < n) d[i] = f2bf(s[i]);
}
// Layer-0 weights: [390,O] -> padded bf16 [3,160,O] (rows 130..159 zero)
__global__ void k_w0pad(unsigned short* __restrict__ dst, const float* __restrict__ W, int O) {
    int i = blockIdx.x * blockDim.x + threadIdx.x;
    int tot = 3 * 160 * O;
    if (i >= tot) return;
    int hop = i / (160 * O), rem = i % (160 * O);
    int row = rem / O, col = rem % O;
    float v = (row < 130) ? W[(size_t)(hop * 130 + row) * O + col] : 0.0f;
    dst[i] = f2bf(v);
}
// z for layer-0 gate gconv: [16384,160] = [x | h0 | 0pad], x = y_prev + dybar_t
__global__ void k_zru0(unsigned short* __restrict__ z, const float* __restrict__ yprev,
                       const float* __restrict__ dyb, int tOff, const float* __restrict__ h0) {
    int i = blockIdx.x * blockDim.x + threadIdx.x;
    const int W = 160;
    if (i >= MTOT * W) return;
    int m = i / W, f = i % W;
    float v;
    if (f < DD) {
        int bb = m / NN, nn = m % NN;
        v = yprev[m * DD + f] + dyb[(size_t)bb * TT * NN * DD + tOff + nn * DD + f];
    } else if (f < DD + HH) {
        v = h0[(size_t)m * HH + (f - DD)];
    } else v = 0.0f;
    z[i] = f2bf(v);
}
// z for layer-0 candidate gconv: [x | r*h0 | 0pad]
__global__ void k_zc0(unsigned short* __restrict__ z, const float* __restrict__ yprev,
                      const float* __restrict__ dyb, int tOff,
                      const float* __restrict__ RU, const float* __restrict__ h0) {
    int i = blockIdx.x * blockDim.x + threadIdx.x;
    const int W = 160;
    if (i >= MTOT * W) return;
    int m = i / W, f = i % W;
    float v;
    if (f < DD) {
        int bb = m / NN, nn = m % NN;
        v = yprev[m * DD + f] + dyb[(size_t)bb * TT * NN * DD + tOff + nn * DD + f];
    } else if (f < DD + HH) {
        int j = f - DD;
        v = RU[(size_t)m * 256 + j] * h0[(size_t)m * HH + j];
    } else v = 0.0f;
    z[i] = f2bf(v);
}
// z for layer-1 gconvs: [h0_new | h1] or [h0_new | r*h1]
__global__ void k_zl1(unsigned short* __restrict__ z, const float* __restrict__ h0n,
                      const float* __restrict__ RU, const float* __restrict__ h1, int useR) {
    int i = blockIdx.x * blockDim.x + threadIdx.x;
    const int W = 256;
    if (i >= MTOT * W) return;
    int m = i / W, f = i % W;
    float v;
    if (f < HH) v = h0n[(size_t)m * HH + f];
    else {
        int j = f - HH;
        v = h1[(size_t)m * HH + j];
        if (useR) v *= RU[(size_t)m * 256 + j];
    }
    z[i] = f2bf(v);
}
// Output head: y = h1 @ W_out + b_out (O=2), also refresh y_prev
__global__ void k_head(const float* __restrict__ h1, const float* __restrict__ Wout,
                       const float* __restrict__ bout, float* __restrict__ out, int tOff,
                       float* __restrict__ yprev) {
    int m = blockIdx.x * blockDim.x + threadIdx.x;
    if (m >= MTOT) return;
    float a0 = bout[0], a1 = bout[1];
#pragma unroll 4
    for (int j = 0; j < HH; ++j) {
        float h = h1[(size_t)m * HH + j];
        a0 += h * Wout[j * 2 + 0];
        a1 += h * Wout[j * 2 + 1];
    }
    int bb = m / NN, nn = m % NN;
    size_t o = (size_t)bb * TT * NN * DD + tOff + (size_t)nn * DD;
    out[o] = a0; out[o + 1] = a1;
    yprev[m * 2] = a0; yprev[m * 2 + 1] = a1;
}

// ---------------------------------------------------------------------------
// Host-side orchestration
// ---------------------------------------------------------------------------
extern "C" void kernel_launch(void* const* d_in, const int* in_sizes, int n_in,
                              void* d_out, int out_size, void* d_ws, size_t ws_size,
                              hipStream_t stream) {
    const float* y0    = (const float*)d_in[0];
    const float* h0in  = (const float*)d_in[1];
    const float* dybar = (const float*)d_in[2];
    const float* topo  = (const float*)d_in[3];
    const float* W_ru0 = (const float*)d_in[4];
    const float* b_ru0 = (const float*)d_in[5];
    const float* W_c0  = (const float*)d_in[6];
    const float* b_c0  = (const float*)d_in[7];
    const float* W_ru1 = (const float*)d_in[8];
    const float* b_ru1 = (const float*)d_in[9];
    const float* W_c1  = (const float*)d_in[10];
    const float* b_c1  = (const float*)d_in[11];
    const float* W_out = (const float*)d_in[12];
    const float* b_out = (const float*)d_in[13];
    float* out = (float*)d_out;

    // Workspace carve (all offsets 256B aligned)
    char* p = (char*)d_ws;
    auto carve = [&](size_t bytes) -> void* {
        void* r = (void*)p;
        p += (bytes + 255) & ~((size_t)255);
        return r;
    };
    float* ws_h0          = (float*)carve(sizeof(float) * MTOT * HH);
    float* ws_h1          = (float*)carve(sizeof(float) * MTOT * HH);
    float* ws_yp          = (float*)carve(sizeof(float) * MTOT * DD);
    float* ws_RU          = (float*)carve(sizeof(float) * MTOT * 256);
    unsigned short* ws_z  = (unsigned short*)carve(2ull * MTOT * 256);
    unsigned short* ws_P  = (unsigned short*)carve(2ull * 3 * MTOT * 256);
    unsigned short* ws_A  = (unsigned short*)carve(2ull * KHOP * NN * NN);
    unsigned short* ws_Wru0 = (unsigned short*)carve(2ull * 3 * 160 * 256);
    unsigned short* ws_Wc0  = (unsigned short*)carve(2ull * 3 * 160 * 128);
    unsigned short* ws_Wru1 = (unsigned short*)carve(2ull * 768 * 256);
    unsigned short* ws_Wc1  = (unsigned short*)carve(2ull * 768 * 128);

    const int THR = 256;
    auto blks = [&](long n) { return (unsigned)((n + THR - 1) / THR); };

    // One-time (per call) prep: state copies + bf16 operand conversion
    k_f32copy<<<blks(MTOT * HH), THR, 0, stream>>>(ws_h0, h0in, MTOT * HH);
    k_f32copy<<<blks(MTOT * HH), THR, 0, stream>>>(ws_h1, h0in + (size_t)MTOT * HH, MTOT * HH);
    k_f32copy<<<blks(MTOT * DD), THR, 0, stream>>>(ws_yp, y0, MTOT * DD);
    k_f2bf<<<blks(KHOP * NN * NN), THR, 0, stream>>>(ws_A, topo, KHOP * NN * NN);
    k_w0pad<<<blks(3 * 160 * 256), THR, 0, stream>>>(ws_Wru0, W_ru0, 256);
    k_w0pad<<<blks(3 * 160 * 128), THR, 0, stream>>>(ws_Wc0, W_c0, 128);
    k_f2bf<<<blks(768 * 256), THR, 0, stream>>>(ws_Wru1, W_ru1, 768 * 256);
    k_f2bf<<<blks(768 * 128), THR, 0, stream>>>(ws_Wc1, W_c1, 768 * 128);

    for (int t = 0; t < TT; ++t) {
        const int tOff = t * NN * DD;

        // ---- Cell 0 (x = y_prev + dybar_t, h = h0) ----
        k_zru0<<<blks((long)MTOT * 160), THR, 0, stream>>>(ws_z, ws_yp, dybar, tOff, ws_h0);
        k_gemm_zw<<<dim3(MTOT / 128, 256 / 64, 3), 256, 0, stream>>>(ws_z, ws_Wru0, ws_P, 160, 256);
        k_gemm_diff<<<dim3(NN / 128, 256 / 64, BB), 256, 0, stream>>>(ws_A, ws_P, b_ru0, ws_RU, ws_h0, 256, 0);
        k_zc0<<<blks((long)MTOT * 160), THR, 0, stream>>>(ws_z, ws_yp, dybar, tOff, ws_RU, ws_h0);
        k_gemm_zw<<<dim3(MTOT / 128, 128 / 64, 3), 256, 0, stream>>>(ws_z, ws_Wc0, ws_P, 160, 128);
        k_gemm_diff<<<dim3(NN / 128, 128 / 64, BB), 256, 0, stream>>>(ws_A, ws_P, b_c0, ws_RU, ws_h0, 128, 1);

        // ---- Cell 1 (x = h0_new, h = h1) ----
        k_zl1<<<blks((long)MTOT * 256), THR, 0, stream>>>(ws_z, ws_h0, ws_RU, ws_h1, 0);
        k_gemm_zw<<<dim3(MTOT / 128, 256 / 64, 3), 256, 0, stream>>>(ws_z, ws_Wru1, ws_P, 256, 256);
        k_gemm_diff<<<dim3(NN / 128, 256 / 64, BB), 256, 0, stream>>>(ws_A, ws_P, b_ru1, ws_RU, ws_h1, 256, 0);
        k_zl1<<<blks((long)MTOT * 256), THR, 0, stream>>>(ws_z, ws_h0, ws_RU, ws_h1, 1);
        k_gemm_zw<<<dim3(MTOT / 128, 128 / 64, 3), 256, 0, stream>>>(ws_z, ws_Wc1, ws_P, 256, 128);
        k_gemm_diff<<<dim3(NN / 128, 128 / 64, BB), 256, 0, stream>>>(ws_A, ws_P, b_c1, ws_RU, ws_h1, 128, 1);

        // ---- Head ----
        k_head<<<blks(MTOT), THR, 0, stream>>>(ws_h1, W_out, b_out, out, tOff, ws_yp);
    }
}
